// TransformerBlock_66314295050600
// MI455X (gfx1250) — matrix operand order repaired and checked by reading, not compiled, not run on hardware
//
#include <hip/hip_runtime.h>

// Problem constants (match reference)
#define B_  2
#define T_  2048
#define D_  1024
#define H_  16
#define HD_ 64
#define FF_ 4096
#define BT_ (B_ * T_)   // 4096 rows

typedef __bf16 bf16_t;
typedef __attribute__((ext_vector_type(16))) __bf16 v16bf;
typedef __attribute__((ext_vector_type(8)))  __bf16 v8bf;
typedef __attribute__((ext_vector_type(8)))  float  v8f;

union FragBF { v16bf v; v8bf h[2]; };

__device__ inline v8f zero8() {
  v8f z;
#pragma unroll
  for (int i = 0; i < 8; i++) z[i] = 0.f;
  return z;
}

__device__ inline v8f wmma_bf16(v16bf a, v16bf b, v8f c) {
  // D = A(16x32 bf16) * B(32x16 bf16) + C(16x16 f32)
  return __builtin_amdgcn_wmma_f32_16x16x32_bf16(
      /*neg_a=*/false, a, /*neg_b=*/false, b,
      /*c_mod=*/(short)0, c, /*reuse_a=*/false, /*reuse_b=*/false);
}

// A-fragment: 16x32 tile from row-major [16, >=k0+32] source, row = lane&15.
// ISA layout: lanes 0-15 hold K = e(0..7), 16+e(8..15); lanes 16-31 offset +8.
__device__ inline v16bf load_frag_mk(const bf16_t* p0, int ld, int lane) {
  const bf16_t* p = p0 + (lane & 15) * ld + ((lane >> 4) << 3);
  FragBF f;
  f.h[0] = *(const v8bf*)(p);        // e=0..7  -> K = 8*hi + e
  f.h[1] = *(const v8bf*)(p + 16);   // e=8..15 -> K = 8*hi + 16 + (e-8)
  return f.v;
}

// B-fragment: 32x16 tile supplied as row-major [N=16, K] ("transposed B"),
// col n = lane&15; K = 8*(lane>>4) + e (e<8), 16 + 8*(lane>>4) + (e-8) (e>=8).
__device__ inline v16bf load_frag_nk(const bf16_t* p0, int ld, int lane) {
  const bf16_t* p = p0 + (lane & 15) * ld + ((lane >> 4) << 3);
  FragBF f;
  f.h[0] = *(const v8bf*)(p);
  f.h[1] = *(const v8bf*)(p + 16);
  return f.v;
}

// ---------------- weight transpose + fp32 -> bf16 ----------------
// W: [K,N] row-major  ->  WT: [N,K] row-major bf16
__global__ __launch_bounds__(256) void convertT(const float* __restrict__ W,
                                                bf16_t* __restrict__ WT,
                                                int K, int N) {
  long e = (long)blockIdx.x * 256 + threadIdx.x;
  if (e >= (long)K * N) return;
  int n = (int)(e / K);
  int k = (int)(e % K);
  WT[e] = (bf16_t)W[(long)k * N + n];
}

// ---------------- LayerNorm (per row of 1024), fp32 + bf16 outputs ----------
__global__ __launch_bounds__(256) void ln_kernel(const float* __restrict__ x,
                                                 const float* __restrict__ g,
                                                 const float* __restrict__ b,
                                                 float* __restrict__ yf,
                                                 bf16_t* __restrict__ yh) {
  const int row = blockIdx.x;
  const int tid = threadIdx.x;
  const float* xr = x + (long)row * D_;
  float v4[4];
  float s = 0.f, s2 = 0.f;
#pragma unroll
  for (int i = 0; i < 4; i++) {
    float v = xr[tid + 256 * i];
    v4[i] = v;
    s += v;
    s2 += v * v;
  }
  __shared__ float rs[256], rq[256];
  rs[tid] = s; rq[tid] = s2;
  __syncthreads();
  for (int o = 128; o > 0; o >>= 1) {
    if (tid < o) { rs[tid] += rs[tid + o]; rq[tid] += rq[tid + o]; }
    __syncthreads();
  }
  const float mean = rs[0] * (1.f / D_);
  const float var  = rq[0] * (1.f / D_) - mean * mean;
  const float rstd = rsqrtf(var + 1e-5f);
#pragma unroll
  for (int i = 0; i < 4; i++) {
    int c = tid + 256 * i;
    float v = (v4[i] - mean) * rstd * g[c] + b[c];
    yf[(long)row * D_ + c] = v;
    yh[(long)row * D_ + c] = (bf16_t)v;
  }
}

// ---------------- WMMA GEMM: C[M,N] = A[M,K] * BT[N,K]^T + bias ------------
// 8 waves/block, each wave: 32 rows x 64 cols (2x4 accumulators; B frags
// shared by both 16-row halves -> 8 WMMAs per 12 b128 loads per K-step).
// MODE 0: bf16 row-major out (optional ReLU)
// MODE 1: bf16 out in Q/K layout [B,H,T,HD]
// MODE 2: bf16 out in V^T layout [B,H,HD,T]
// MODE 3: fp32 out = val + resid (row-major)
template <int MODE, bool RELU>
__global__ __launch_bounds__(256) void gemm_bf16_wmma(
    const bf16_t* __restrict__ A, const bf16_t* __restrict__ BT,
    const float* __restrict__ bias, const float* __restrict__ resid,
    bf16_t* __restrict__ outh, float* __restrict__ outf,
    int M, int N, int K) {
  const int lane = threadIdx.x & 31;
  const int wave = threadIdx.x >> 5;
  const int hi = lane >> 4, lo = lane & 15;
  const int m0 = blockIdx.y * 256 + wave * 32;
  const int n0 = blockIdx.x * 64;

  v8f acc[2][4];
#pragma unroll
  for (int i = 0; i < 2; i++)
#pragma unroll
    for (int j = 0; j < 4; j++) acc[i][j] = zero8();

  const bf16_t* Ab0 = A + (long)m0 * K;
  const bf16_t* Ab1 = A + (long)(m0 + 16) * K;
  for (int k0 = 0; k0 < K; k0 += 32) {
    v16bf a0 = load_frag_mk(Ab0 + k0, K, lane);
    v16bf a1 = load_frag_mk(Ab1 + k0, K, lane);
#pragma unroll
    for (int j = 0; j < 4; j++) {
      v16bf b = load_frag_nk(BT + (long)(n0 + 16 * j) * K + k0, K, lane);
      acc[0][j] = wmma_bf16(a0, b, acc[0][j]);
      acc[1][j] = wmma_bf16(a1, b, acc[1][j]);
    }
  }

#pragma unroll
  for (int i = 0; i < 2; i++) {
#pragma unroll
    for (int j = 0; j < 4; j++) {
      const int n = n0 + 16 * j + lo;
      const float bi = bias[n];
#pragma unroll
      for (int r = 0; r < 8; r++) {
        const int m = m0 + 16 * i + r + 8 * hi;
        float v = acc[i][j][r] + bi;
        if (RELU) v = fmaxf(v, 0.f);
        if (MODE == 0) {
          outh[(long)m * N + n] = (bf16_t)v;
        } else if (MODE == 1) {
          int bb = m >> 11, t = m & (T_ - 1), hh = n >> 6, dd = n & (HD_ - 1);
          outh[(((long)(bb * H_ + hh)) * T_ + t) * HD_ + dd] = (bf16_t)v;
        } else if (MODE == 2) {
          int bb = m >> 11, t = m & (T_ - 1), hh = n >> 6, dd = n & (HD_ - 1);
          outh[(((long)(bb * H_ + hh)) * HD_ + dd) * T_ + t] = (bf16_t)v;
        } else {
          outf[(long)m * N + n] = v + resid[(long)m * N + n];
        }
      }
    }
  }
}

// ---------------- flash attention: 1 wave per 16-row q-tile ----------------
// Q,K: [B,H,T,HD] bf16;  Vt: [B,H,HD,T] bf16;  ctx: [B,T,D] bf16
__global__ __launch_bounds__(256) void attn_flash_wmma(
    const bf16_t* __restrict__ Q, const bf16_t* __restrict__ Kk,
    const bf16_t* __restrict__ Vt, bf16_t* __restrict__ ctx) {
  __shared__ __align__(16) bf16_t pbuf[8][16][32];  // per-wave P tile
  const int lane = threadIdx.x & 31;
  const int wave = threadIdx.x >> 5;
  const int hi = lane >> 4, lo = lane & 15;
  const int g = blockIdx.x * 8 + wave;
  const int bh = g >> 7;            // /(T/16)=128 q-tiles per (b,h)
  const int q0 = (g & 127) * 16;

  const bf16_t* Qb = Q  + (long)bh * T_ * HD_ + (long)q0 * HD_;
  const bf16_t* Kb = Kk + (long)bh * T_ * HD_;
  const bf16_t* Vb = Vt + (long)bh * HD_ * T_;

  const v16bf qa0 = load_frag_mk(Qb, HD_, lane);        // d = 0..31
  const v16bf qa1 = load_frag_mk(Qb + 32, HD_, lane);   // d = 32..63

  v8f acc[4];
#pragma unroll
  for (int j = 0; j < 4; j++) acc[j] = zero8();
  float mrow[8], lrow[8];
#pragma unroll
  for (int r = 0; r < 8; r++) { mrow[r] = -1e30f; lrow[r] = 0.f; }

  bf16_t* pb = &pbuf[wave][0][0];

  for (int kt = 0; kt < q0 + 16; kt += 32) {
    // scores: S[16q x 32k], contraction over HD=64 (two K=32 WMMAs per half)
    v8f s0 = zero8(), s1 = zero8();
    s0 = wmma_bf16(qa0, load_frag_nk(Kb + (long)kt * HD_,          HD_, lane), s0);
    s0 = wmma_bf16(qa1, load_frag_nk(Kb + (long)kt * HD_ + 32,     HD_, lane), s0);
    s1 = wmma_bf16(qa0, load_frag_nk(Kb + (long)(kt + 16) * HD_,      HD_, lane), s1);
    s1 = wmma_bf16(qa1, load_frag_nk(Kb + (long)(kt + 16) * HD_ + 32, HD_, lane), s1);

    float p0[8], p1[8], tm[8];
#pragma unroll
    for (int r = 0; r < 8; r++) {
      const int q = q0 + r + 8 * hi;
      float a0 = (kt + lo      <= q) ? s0[r] * 0.125f : -1e30f;  // 1/sqrt(64)
      float a1 = (kt + 16 + lo <= q) ? s1[r] * 0.125f : -1e30f;
      p0[r] = a0; p1[r] = a1;
      tm[r] = fmaxf(a0, a1);
    }
    // row-max across the 16 lanes holding one row (xor masks stay in half)
#pragma unroll
    for (int off = 1; off < 16; off <<= 1) {
#pragma unroll
      for (int r = 0; r < 8; r++) tm[r] = fmaxf(tm[r], __shfl_xor(tm[r], off, 32));
    }
    float al[8], rs[8];
#pragma unroll
    for (int r = 0; r < 8; r++) {
      float mn = fmaxf(mrow[r], tm[r]);
      al[r] = __expf(mrow[r] - mn);
      mrow[r] = mn;
      p0[r] = __expf(p0[r] - mn);
      p1[r] = __expf(p1[r] - mn);
      rs[r] = p0[r] + p1[r];
    }
#pragma unroll
    for (int off = 1; off < 16; off <<= 1) {
#pragma unroll
      for (int r = 0; r < 8; r++) rs[r] += __shfl_xor(rs[r], off, 32);
    }
#pragma unroll
    for (int r = 0; r < 8; r++) lrow[r] = lrow[r] * al[r] + rs[r];
#pragma unroll
    for (int j = 0; j < 4; j++)
#pragma unroll
      for (int r = 0; r < 8; r++) acc[j][r] *= al[r];

    // P (D-layout) -> LDS -> A-layout fragment (cross-lane, wave-private tile)
#pragma unroll
    for (int r = 0; r < 8; r++) {
      pb[(r + 8 * hi) * 32 + lo]      = (bf16_t)p0[r];
      pb[(r + 8 * hi) * 32 + 16 + lo] = (bf16_t)p1[r];
    }
    asm volatile("s_wait_dscnt 0" ::: "memory");  // cross-lane RAW on pbuf
    v16bf pa = load_frag_mk(pb, 32, lane);
    // out += P(16x32) * V(32x64): Vt is [N=d][K=t] row-major
#pragma unroll
    for (int j = 0; j < 4; j++)
      acc[j] = wmma_bf16(pa, load_frag_nk(Vb + (long)(16 * j) * T_ + kt, T_, lane), acc[j]);
    asm volatile("s_wait_dscnt 0" ::: "memory");  // WAR before next-iter stores
  }

  const int bb = bh >> 4, hh = bh & 15;
#pragma unroll
  for (int j = 0; j < 4; j++) {
#pragma unroll
    for (int r = 0; r < 8; r++) {
      int t = q0 + r + 8 * hi;
      int d = 16 * j + lo;
      float o = acc[j][r] / lrow[r];
      ctx[((long)(bb * T_ + t)) * D_ + hh * HD_ + d] = (bf16_t)o;
    }
  }
}

// --------------------------------- launch ---------------------------------
extern "C" void kernel_launch(void* const* d_in, const int* in_sizes, int n_in,
                              void* d_out, int out_size, void* d_ws, size_t ws_size,
                              hipStream_t stream) {
  (void)in_sizes; (void)n_in; (void)out_size; (void)ws_size;
  const float* x   = (const float*)d_in[0];
  const float* Wq  = (const float*)d_in[1];
  const float* bq  = (const float*)d_in[2];
  const float* Wk  = (const float*)d_in[3];
  const float* bk  = (const float*)d_in[4];
  const float* Wv  = (const float*)d_in[5];
  const float* bv  = (const float*)d_in[6];
  const float* Wo  = (const float*)d_in[7];
  const float* bo  = (const float*)d_in[8];
  const float* W1  = (const float*)d_in[9];
  const float* b1  = (const float*)d_in[10];
  const float* W2  = (const float*)d_in[11];
  const float* b2  = (const float*)d_in[12];
  const float* g1  = (const float*)d_in[13];
  const float* be1 = (const float*)d_in[14];
  const float* g2  = (const float*)d_in[15];
  const float* be2 = (const float*)d_in[16];

  char* p = (char*)d_ws;
  auto take = [&](size_t bytes) { char* q = p; p += (bytes + 255) & ~(size_t)255; return q; };

  bf16_t* WqT  = (bf16_t*)take((size_t)D_ * D_ * 2);
  bf16_t* WkT  = (bf16_t*)take((size_t)D_ * D_ * 2);
  bf16_t* WvT  = (bf16_t*)take((size_t)D_ * D_ * 2);
  bf16_t* WoT  = (bf16_t*)take((size_t)D_ * D_ * 2);
  bf16_t* W1T  = (bf16_t*)take((size_t)D_ * FF_ * 2);   // [FF, D]
  bf16_t* W2T  = (bf16_t*)take((size_t)FF_ * D_ * 2);   // [D, FF]
  float*  xn1f = (float*) take((size_t)BT_ * D_ * 4);
  bf16_t* xn1h = (bf16_t*)take((size_t)BT_ * D_ * 2);
  bf16_t* Qb   = (bf16_t*)take((size_t)BT_ * D_ * 2);   // [B,H,T,HD]
  bf16_t* Kb   = (bf16_t*)take((size_t)BT_ * D_ * 2);   // [B,H,T,HD]
  bf16_t* Vtb  = (bf16_t*)take((size_t)BT_ * D_ * 2);   // [B,H,HD,T]
  bf16_t* ctx  = (bf16_t*)take((size_t)BT_ * D_ * 2);   // [B,T,D]
  float*  x2f  = (float*) take((size_t)BT_ * D_ * 4);
  float*  x3f  = (float*) take((size_t)BT_ * D_ * 4);
  bf16_t* x3h  = (bf16_t*)take((size_t)BT_ * D_ * 2);
  bf16_t* hb   = (bf16_t*)take((size_t)BT_ * FF_ * 2);  // [BT, FF]

  const dim3 blk(256);

  // bf16 transposed weights
  convertT<<<(D_ * D_) / 256, blk, 0, stream>>>(Wq, WqT, D_, D_);
  convertT<<<(D_ * D_) / 256, blk, 0, stream>>>(Wk, WkT, D_, D_);
  convertT<<<(D_ * D_) / 256, blk, 0, stream>>>(Wv, WvT, D_, D_);
  convertT<<<(D_ * D_) / 256, blk, 0, stream>>>(Wo, WoT, D_, D_);
  convertT<<<(D_ * FF_) / 256, blk, 0, stream>>>(W1, W1T, D_, FF_);
  convertT<<<(FF_ * D_) / 256, blk, 0, stream>>>(W2, W2T, FF_, D_);

  // x1 = LN(x)
  ln_kernel<<<BT_, blk, 0, stream>>>(x, g1, be1, xn1f, xn1h);

  // Q/K/V projections  (grid: N/64 x M/256)
  const dim3 gD(D_ / 64, BT_ / 256);   // (16, 16)
  gemm_bf16_wmma<1, false><<<gD, blk, 0, stream>>>(xn1h, WqT, bq, nullptr, Qb,  nullptr, BT_, D_, D_);
  gemm_bf16_wmma<1, false><<<gD, blk, 0, stream>>>(xn1h, WkT, bk, nullptr, Kb,  nullptr, BT_, D_, D_);
  gemm_bf16_wmma<2, false><<<gD, blk, 0, stream>>>(xn1h, WvT, bv, nullptr, Vtb, nullptr, BT_, D_, D_);

  // causal flash attention
  attn_flash_wmma<<<(B_ * H_ * (T_ / 16)) / 8, blk, 0, stream>>>(Qb, Kb, Vtb, ctx);

  // x2 = x1 + ctx @ Wo + bo
  gemm_bf16_wmma<3, false><<<gD, blk, 0, stream>>>(ctx, WoT, bo, xn1f, nullptr, x2f, BT_, D_, D_);

  // x3 = LN(x2)
  ln_kernel<<<BT_, blk, 0, stream>>>(x2f, g2, be2, x3f, x3h);

  // FFN: h = relu(x3 @ W1 + b1);  out = x3 + h @ W2 + b2
  const dim3 gFF(FF_ / 64, BT_ / 256); // (64, 16)
  gemm_bf16_wmma<0, true ><<<gFF, blk, 0, stream>>>(x3h, W1T, b1, nullptr, hb, nullptr, BT_, FF_, D_);
  gemm_bf16_wmma<3, false><<<gD,  blk, 0, stream>>>(hb,  W2T, b2, x3f, nullptr, (float*)d_out, BT_, D_, FF_);
}
